// MultiHeadAttention_36885179138123
// MI455X (gfx1250) — matrix operand order repaired and checked by reading, not compiled, not run on hardware
//
#include <hip/hip_runtime.h>
#include <cstdint>
#include <cstddef>

// ---------------------------------------------------------------------------
// MHA  B=2, S=2048, D=1024, H=16, DH=64  on gfx1250 (wave32, WMMA).
// GEMMs: v_wmma_f32_16x16x32_f16; A tile staged by TDM (tensor_load_to_lds,
// TENSORcnt), trans-B tile by GLOBAL_LOAD_ASYNC_TO_LDS_B128 (ASYNCcnt),
// nn-B tile by reg transpose-scatter. Double-buffered LDS.
// attn [B,H,S,S] fp32 (536 MB) materialized in d_out => HBM-bound;
// logits + softmax use non-temporal load/store (tensor is 2.8x L2).
// ---------------------------------------------------------------------------

typedef _Float16 half_t;
typedef __attribute__((ext_vector_type(16))) _Float16 v16h;
typedef __attribute__((ext_vector_type(8)))  _Float16 v8h;
typedef __attribute__((ext_vector_type(4)))  _Float16 v4h;
typedef __attribute__((ext_vector_type(8)))  float    v8f;
typedef __attribute__((ext_vector_type(4)))  float    v4f;
typedef __attribute__((ext_vector_type(4)))  int      v4i;
typedef __attribute__((ext_vector_type(8)))  int      v8i;
typedef __attribute__((ext_vector_type(4)))  unsigned v4u;

#define BLK_K 32
#define APAD  8          // row stride 40 halves = 80 B (multiple of 16 B)

// ---- async global->LDS (GLOBAL_LOAD_ASYNC_TO_LDS_B128, ASYNCcnt) ----
#if __has_builtin(__builtin_amdgcn_global_load_async_to_lds_b128) && \
    __has_builtin(__builtin_amdgcn_s_wait_asynccnt)
#define ASYNC_LDS 1
#else
#define ASYNC_LDS 0
#endif

// ---- Tensor Data Mover (TENSOR_LOAD_TO_LDS, TENSORcnt) ----
#if __has_builtin(__builtin_amdgcn_tensor_load_to_lds) && \
    __has_builtin(__builtin_amdgcn_s_wait_tensorcnt)
#define HAVE_TDM 1
#else
#define HAVE_TDM 0
#endif

typedef __attribute__((address_space(1))) v4i    as1_v4i;
typedef __attribute__((address_space(3))) v4i    as3_v4i;
typedef __attribute__((address_space(3))) half_t as3_half;

__device__ __forceinline__ void cp16_to_lds(half_t* lds, const half_t* g)
{
#if ASYNC_LDS
  __builtin_amdgcn_global_load_async_to_lds_b128((as1_v4i*)g, (as3_v4i*)lds, 0, 0);
#else
  *(v8h*)lds = *(const v8h*)g;
#endif
}
__device__ __forceinline__ void async_join()
{
#if ASYNC_LDS
  __builtin_amdgcn_s_wait_asynccnt(0);
#endif
}

#if HAVE_TDM
// D# for a 2-D tile: rows x 32 halves, row stride lda halves, dest LDS rows
// padded 64B data + 16B pad (pad_interval=3 -> 16 DW, pad_amount=3 -> 4 DW).
__device__ __forceinline__ void tdm_load_tile(const half_t* gbase,
                                              unsigned lds_off,
                                              int lda, int rows)
{
  const unsigned long long ga = (unsigned long long)(uintptr_t)gbase;
  v4u g0;
  g0[0] = 1u;                                   // count=1, user descriptor
  g0[1] = lds_off;                              // lds_addr (bytes)
  g0[2] = (unsigned)ga;                         // global_addr[31:0]
  g0[3] = (unsigned)((ga >> 32) & 0x01FFFFFFu)  // global_addr[56:32]
        | (2u << 30);                           // type = 2 ("image")
  v8i g1;
  g1[0] = (int)((1u << 16)                      // data_size = 1 (2 bytes)
        | (1u << 20)                            // pad_enable
        | (3u << 22)                            // pad_interval: 16 DWORDs
        | (3u << 25));                          // pad_amount:   4 DWORDs
  g1[1] = (int)(((unsigned)lda & 0xFFFFu) << 16);            // tensor_dim0 lo16
  g1[2] = (int)((((unsigned)lda >> 16) & 0xFFFFu)            // tensor_dim0 hi16
        | (((unsigned)rows & 0xFFFFu) << 16));               // tensor_dim1 lo16
  g1[3] = (int)(32u << 16);                     // tensor_dim1 hi16=0, tile_dim0=32
  g1[4] = rows;                                 // tile_dim1=rows, tile_dim2=0
  g1[5] = lda;                                  // tensor_dim0_stride[31:0]
  g1[6] = 0;                                    // stride hi, dim1_stride lo
  g1[7] = 0;
  v4i z4 = {0, 0, 0, 0};                        // 2-D: groups 2/3 unused
  v8i z8 = {0, 0, 0, 0, 0, 0, 0, 0};
  __builtin_amdgcn_tensor_load_to_lds(g0, g1, z4, z4, z8, 0);
}
#endif

struct GemmP {
  const half_t* A;       // M x K row-major (f16)
  const half_t* B;       // nn: K x N row-major; nt: N x K row-major (f16)
  const float*  bias;    // [N] or null
  const float*  mask;    // [M x ldmask] or null (added as -1e9*mask)
  float*        C32;     // optional fp32 out
  half_t*       C16;     // optional f16 out
  int M, N, K;
  int lda, ldb, ldc, ldmask;
  float alpha;
  int Hdiv;              // z = b*Hdiv + h (1 => unbatched)
  long long sAb, sAh, sBb, sBh, sCb, sCh;
};

// BM x BN block tile, 8 waves, each wave a 32x32 tile (4 wmma accumulators).
// Requires M % BM == 0, N % BN == 0, K % 32 == 0 (guaranteed by launches).
template<int BM, int BN, bool TRANS_B, bool HAS_BIAS, bool HAS_MASK,
         bool OUT32, bool OUT16, bool NT_C>
__global__ __launch_bounds__(256)
void wmma_gemm(GemmP p)
{
  static_assert(BM * BN == 8192, "8 waves x 32x32");
  constexpr int WCOLS = BN / 32;
  constexpr int ACH = BM / 64;     // 16B chunks per thread for A tile (fallback)
  constexpr int BCH = BN / 64;     // 16B chunks per thread for trans-B tile
  constexpr int BNN = BN / 8;      // halves per thread along n for nn-B tile

  __shared__ __align__(16) half_t As[2][BM][BLK_K + APAD];
  __shared__ __align__(16) half_t Bs[2][BN][BLK_K + APAD];   // n-major, k contiguous

  const int tid  = threadIdx.x;
  const int lane = tid & 31;
  const int wave = tid >> 5;
  const int wm   = wave / WCOLS;
  const int wn   = wave % WCOLS;

  const int z  = blockIdx.z;
  const int bz = z / p.Hdiv;
  const int hz = z % p.Hdiv;

  const half_t* A = p.A + (long long)bz * p.sAb + (long long)hz * p.sAh;
  const half_t* B = p.B + (long long)bz * p.sBb + (long long)hz * p.sBh;
  const long long cOff = (long long)bz * p.sCb + (long long)hz * p.sCh;

  const int m0 = blockIdx.y * BM;
  const int n0 = blockIdx.x * BN;
  const int nTiles = p.K / BLK_K;

  v8f acc[2][2] = {};
  half_t breg[BNN];                 // staging for nn-B transpose scatter

  auto loadA = [&](int kt, int buf) {
#if HAVE_TDM
    if (wave == 0) {                // one TDM op stages the whole BM x 32 tile
      const unsigned lds_off =
          (unsigned)(unsigned long long)(as3_half*)&As[buf][0][0];
      tdm_load_tile(A + (size_t)m0 * p.lda + kt * BLK_K, lds_off, p.lda, BM);
    }
#else
    #pragma unroll
    for (int c = 0; c < ACH; ++c) {
      const int cid = tid + c * 256;
      const int row = cid >> 2;
      const int col = (cid & 3) * 8;
      cp16_to_lds(&As[buf][row][col],
                  A + (size_t)(m0 + row) * p.lda + kt * BLK_K + col);
    }
#endif
  };
  auto loadBtrans = [&](int kt, int buf) {
    #pragma unroll
    for (int c = 0; c < BCH; ++c) {
      const int cid = tid + c * 256;
      const int n  = cid >> 2;
      const int kk = (cid & 3) * 8;
      cp16_to_lds(&Bs[buf][n][kk],
                  B + (size_t)(n0 + n) * p.ldb + kt * BLK_K + kk);
    }
  };
  auto gloadBnn = [&](int kt) {
    const int kk = tid >> 3;
    const int nn = (tid & 7) * BNN;
    const half_t* gb = B + (size_t)(kt * BLK_K + kk) * p.ldb + (n0 + nn);
    #pragma unroll
    for (int c = 0; c < BNN / 8; ++c)
      *(v8h*)&breg[c * 8] = *(const v8h*)(gb + c * 8);
    if (kt + 1 < nTiles)
      __builtin_prefetch(gb + (size_t)BLK_K * p.ldb, 0, 3);   // near prefetch
  };
  auto scatterBnn = [&](int buf) {
    const int kk = tid >> 3;
    const int nn = (tid & 7) * BNN;
    #pragma unroll
    for (int j = 0; j < BNN; ++j) Bs[buf][nn + j][kk] = breg[j];
  };
  auto tile_join = [&]() {          // all staging traffic for a tile complete
#if HAVE_TDM
    if (wave == 0) __builtin_amdgcn_s_wait_tensorcnt(0);
#endif
    async_join();
  };

  // ---- prologue: tile 0 into buffer 0 ----
  loadA(0, 0);
  if (TRANS_B) loadBtrans(0, 0);
  else { gloadBnn(0); scatterBnn(0); }
  tile_join();
  __syncthreads();

  int cur = 0;
  for (int kt = 0; kt < nTiles; ++kt) {
    const int nxt = cur ^ 1;
    const bool more = (kt + 1 < nTiles);

    // issue next tile's DMA/global traffic before computing (overlap)
    if (more) {
      loadA(kt + 1, nxt);
      if (TRANS_B) loadBtrans(kt + 1, nxt);
      else gloadBnn(kt + 1);
    }

    // ---- 4 WMMAs on current buffers ----
    #pragma unroll
    for (int i = 0; i < 2; ++i) {
      // A fragment (16x32 f16): lanes0-15 K 0..7 & 16..23; lanes16-31 K 8..15 & 24..31
      const int ar = wm * 32 + i * 16 + (lane & 15);
      const int ak = (lane >> 4) * 8;
      v8h a0 = *(const v8h*)&As[cur][ar][ak];
      v8h a1 = *(const v8h*)&As[cur][ar][ak + 16];
      v16h a;
      #pragma unroll
      for (int t = 0; t < 8; ++t) { a[t] = a0[t]; a[t + 8] = a1[t]; }

      #pragma unroll
      for (int j = 0; j < 2; ++j) {
        // B fragment (32x16 f16): lane n = lane&15; lanes0-15 K 0..7 & 16..23; lanes16-31 K 8..15 & 24..31
        const int bcn = wn * 32 + j * 16 + (lane & 15);
        const int bk  = (lane >> 4) * 8;
        v8h c0 = *(const v8h*)&Bs[cur][bcn][bk];
        v8h c1 = *(const v8h*)&Bs[cur][bcn][bk + 16];
        v16h b;
        #pragma unroll
        for (int t = 0; t < 8; ++t) { b[t] = c0[t]; b[t + 8] = c1[t]; }

        acc[i][j] = __builtin_amdgcn_wmma_f32_16x16x32_f16(
            false, a, false, b, (short)0, acc[i][j], false, false);
      }
    }

    if (more) {
      if (!TRANS_B) scatterBnn(nxt);   // writes buffer 'nxt': no conflict
      tile_join();                     // next tile's LDS writes landed
    }
    __syncthreads();
    cur = nxt;
  }

  // ---- epilogue: lane&15 -> N, VGPR r -> M = (lane>>4)*8 + r ----
  #pragma unroll
  for (int i = 0; i < 2; ++i) {
    #pragma unroll
    for (int j = 0; j < 2; ++j) {
      const int mBase = m0 + wm * 32 + i * 16 + (lane >> 4) * 8;
      const int n = n0 + wn * 32 + j * 16 + (lane & 15);
      float bv = 0.f;
      if (HAS_BIAS) bv = p.bias[n];
      #pragma unroll
      for (int r = 0; r < 8; ++r) {
        const int m = mBase + r;
        float v = acc[i][j][r] * p.alpha;
        if (HAS_BIAS) v += bv;
        if (HAS_MASK) v += -1e9f * p.mask[(size_t)m * p.ldmask + n];
        const long long idx = cOff + (long long)m * p.ldc + n;
        if (OUT32) {
          if (NT_C) __builtin_nontemporal_store(v, &p.C32[idx]);
          else      p.C32[idx] = v;
        }
        if (OUT16) p.C16[idx] = (half_t)v;
      }
    }
  }
}

// ---------------------------------------------------------------------------
// vectorized f32 -> f16: 8 elements/thread, n must be a multiple of 2048
__global__ __launch_bounds__(256)
void f32_to_f16(const float* __restrict__ in, half_t* __restrict__ out)
{
  const size_t i8 = ((size_t)blockIdx.x * 256 + threadIdx.x) * 8;
  const v4f* in4 = (const v4f*)(in + i8);
  v4f a = in4[0];
  v4f b = in4[1];
  v8h h;
  #pragma unroll
  for (int t = 0; t < 4; ++t) { h[t] = (half_t)a[t]; h[t + 4] = (half_t)b[t]; }
  *(v8h*)(out + i8) = h;
}

// in-place row softmax over S=2048; emits f16 copy. Register-resident,
// non-temporal b128 traffic (attn is 2.8x L2).
#define SEQ 2048
__global__ __launch_bounds__(256)
void softmax_row(float* __restrict__ attn, half_t* __restrict__ attn16)
{
  __shared__ float red[256];
  const int tid = threadIdx.x;
  v4f* p4 = (v4f*)(attn + (size_t)blockIdx.x * SEQ);

  v4f r0 = __builtin_nontemporal_load(&p4[tid]);
  v4f r1 = __builtin_nontemporal_load(&p4[tid + 256]);

  float mx = fmaxf(fmaxf(fmaxf(r0[0], r0[1]), fmaxf(r0[2], r0[3])),
                   fmaxf(fmaxf(r1[0], r1[1]), fmaxf(r1[2], r1[3])));
  red[tid] = mx; __syncthreads();
  #pragma unroll
  for (int s = 128; s > 0; s >>= 1) {
    if (tid < s) red[tid] = fmaxf(red[tid], red[tid + s]);
    __syncthreads();
  }
  mx = red[0]; __syncthreads();

  float sum = 0.f;
  #pragma unroll
  for (int t = 0; t < 4; ++t) { r0[t] = __expf(r0[t] - mx); sum += r0[t]; }
  #pragma unroll
  for (int t = 0; t < 4; ++t) { r1[t] = __expf(r1[t] - mx); sum += r1[t]; }
  red[tid] = sum; __syncthreads();
  #pragma unroll
  for (int s = 128; s > 0; s >>= 1) {
    if (tid < s) red[tid] += red[tid + s];
    __syncthreads();
  }
  const float inv = 1.f / red[0];
  r0 *= inv; r1 *= inv;

  __builtin_nontemporal_store(r0, &p4[tid]);
  __builtin_nontemporal_store(r1, &p4[tid + 256]);

  half_t* p16 = attn16 + (size_t)blockIdx.x * SEQ;
  v4h h0, h1;
  #pragma unroll
  for (int t = 0; t < 4; ++t) { h0[t] = (half_t)r0[t]; h1[t] = (half_t)r1[t]; }
  __builtin_nontemporal_store(h0, &((v4h*)p16)[tid]);
  __builtin_nontemporal_store(h1, &((v4h*)p16)[tid + 256]);
}

// ---------------------------------------------------------------------------
extern "C" void kernel_launch(void* const* d_in, const int* in_sizes, int n_in,
                              void* d_out, int out_size, void* d_ws, size_t ws_size,
                              hipStream_t stream)
{
  constexpr int Bb = 2, S = 2048, D = 1024, H = 16, DH = 64;
  (void)in_sizes; (void)n_in; (void)out_size; (void)ws_size;

  const float* x    = (const float*)d_in[0];
  const float* mask = (const float*)d_in[1];
  const float* Wq   = (const float*)d_in[2];
  const float* bq   = (const float*)d_in[3];
  const float* Wk   = (const float*)d_in[4];
  const float* bk   = (const float*)d_in[5];
  const float* Wv   = (const float*)d_in[6];
  const float* bv   = (const float*)d_in[7];
  const float* Wo   = (const float*)d_in[8];
  const float* bo   = (const float*)d_in[9];

  float* outP  = (float*)d_out;                 // [B,S,D] fp32
  float* attnP = outP + (size_t)Bb * S * D;     // [B,H,S,S] fp32

  half_t* w = (half_t*)d_ws;
  half_t* x16    = w; w += (size_t)Bb * S * D;
  half_t* Wq16   = w; w += (size_t)D * D;
  half_t* Wk16   = w; w += (size_t)D * D;
  half_t* Wv16   = w; w += (size_t)D * D;
  half_t* Wo16   = w; w += (size_t)D * D;
  half_t* Q16    = w; w += (size_t)Bb * S * D;
  half_t* K16    = w; w += (size_t)Bb * S * D;
  half_t* V16    = w; w += (size_t)Bb * S * D;
  half_t* ctx16  = w; w += (size_t)Bb * S * D;
  half_t* attn16 = w; w += (size_t)Bb * H * S * S;

  auto cvt = [&](const float* src, half_t* dst, size_t n) {
    f32_to_f16<<<(unsigned)(n / 2048), 256, 0, stream>>>(src, dst);
  };
  cvt(x,  x16,  (size_t)Bb * S * D);
  cvt(Wq, Wq16, (size_t)D * D);
  cvt(Wk, Wk16, (size_t)D * D);
  cvt(Wv, Wv16, (size_t)D * D);
  cvt(Wo, Wo16, (size_t)D * D);

  // ---- QKV projections: [4096,1024] = x16 @ W16 + b -> f16 ----
  {
    GemmP p{};
    p.A = x16; p.M = Bb * S; p.N = D; p.K = D;
    p.lda = D; p.ldb = D; p.ldc = D; p.ldmask = 0;
    p.alpha = 1.f; p.Hdiv = 1;
    p.sAb = p.sAh = p.sBb = p.sBh = p.sCb = p.sCh = 0;
    dim3 grid(D / 128, (Bb * S) / 64, 1);

    p.B = Wq16; p.bias = bq; p.C16 = Q16;
    wmma_gemm<64, 128, false, true, false, false, true, false><<<grid, 256, 0, stream>>>(p);
    p.B = Wk16; p.bias = bk; p.C16 = K16;
    wmma_gemm<64, 128, false, true, false, false, true, false><<<grid, 256, 0, stream>>>(p);
    p.B = Wv16; p.bias = bv; p.C16 = V16;
    wmma_gemm<64, 128, false, true, false, false, true, false><<<grid, 256, 0, stream>>>(p);
  }

  // ---- logits: attn[b,h] = Q @ K^T * (1/8) - 1e9*mask  (fp32, NT stores) ----
  {
    GemmP p{};
    p.A = Q16; p.B = K16; p.mask = mask; p.C32 = attnP;
    p.M = S; p.N = S; p.K = DH;
    p.lda = D; p.ldb = D; p.ldc = S; p.ldmask = S;
    p.alpha = 0.125f;
    p.Hdiv = H;
    p.sAb = (long long)S * D; p.sAh = DH;
    p.sBb = (long long)S * D; p.sBh = DH;
    p.sCb = (long long)H * S * S; p.sCh = (long long)S * S;
    dim3 grid(S / 128, S / 64, Bb * H);
    wmma_gemm<64, 128, true, false, true, true, false, true><<<grid, 256, 0, stream>>>(p);
  }

  // ---- softmax in place; emit f16 attn ----
  softmax_row<<<Bb * H * S, 256, 0, stream>>>(attnP, attn16);

  // ---- ctx[b,h] = attn16 @ V16 -> ctx16 as [B,S,D]; tile 128x64 (N=64) ----
  {
    GemmP p{};
    p.A = attn16; p.B = V16; p.C16 = ctx16;
    p.M = S; p.N = DH; p.K = S;
    p.lda = S; p.ldb = D; p.ldc = D; p.ldmask = 0;
    p.alpha = 1.f; p.Hdiv = H;
    p.sAb = (long long)H * S * S; p.sAh = (long long)S * S;
    p.sBb = (long long)S * D; p.sBh = DH;
    p.sCb = (long long)S * D; p.sCh = DH;
    dim3 grid(DH / 64, S / 128, Bb * H);
    wmma_gemm<128, 64, false, false, false, false, true, false><<<grid, 256, 0, stream>>>(p);
  }

  // ---- out = ctx16 @ Wo16 + bo -> fp32 d_out ----
  {
    GemmP p{};
    p.A = ctx16; p.B = Wo16; p.bias = bo; p.C32 = outP;
    p.M = Bb * S; p.N = D; p.K = D;
    p.lda = D; p.ldb = D; p.ldc = D; p.ldmask = 0;
    p.alpha = 1.f; p.Hdiv = 1;
    p.sAb = p.sAh = p.sBb = p.sBh = p.sCb = p.sCh = 0;
    dim3 grid(D / 128, (Bb * S) / 64, 1);
    wmma_gemm<64, 128, false, true, false, true, false, false><<<grid, 256, 0, stream>>>(p);
  }
}
